// BipartiteGCN_46033459479165
// MI455X (gfx1250) — matrix operand order repaired and checked by reading, not compiled, not run on hardware
//
#include <hip/hip_runtime.h>
#include <hip/hip_bf16.h>

// ---- model dims ----
#define NN      20000
#define SS      16
#define EE      160000
#define NMPm    2
#define HH      4
#define OO      64
#define D_FEAT  64
#define E_DIM   32
#define PREP    128
#define NCLS    8
#define BB      4096
#define D1      256          // H*O
#define SL      (BB*SS)      // 65536 gathered edge slots

typedef __attribute__((ext_vector_type(16))) __bf16 v16bf;
typedef __attribute__((ext_vector_type(8)))  float  v8f;
typedef unsigned int v4u __attribute__((ext_vector_type(4)));
typedef int          v8i __attribute__((ext_vector_type(8)));
typedef int          v4i __attribute__((ext_vector_type(4)));

// ---------------------------------------------------------------------------
// f32 -> bf16 elementwise convert
// ---------------------------------------------------------------------------
__global__ void k_conv(const float* __restrict__ src, __bf16* __restrict__ dst, int n) {
  int i = blockIdx.x * 256 + threadIdx.x;
  if (i < n) dst[i] = (__bf16)src[i];
}

// ---------------------------------------------------------------------------
// weight transpose+convert: src f32 [nmat][K][Nc] -> dst bf16 [nmat][Nc][K]
// ---------------------------------------------------------------------------
__global__ void k_wt(const float* __restrict__ src, __bf16* __restrict__ dst,
                     int K, int Nc, int total) {
  int i = blockIdx.x * 256 + threadIdx.x;
  if (i >= total) return;
  int mk  = K * Nc;
  int mat = i / mk;
  int rem = i - mat * mk;
  int r = rem / Nc, c = rem - r * Nc;
  dst[(size_t)mat * mk + (size_t)c * K + r] = (__bf16)src[i];
}

// ---------------------------------------------------------------------------
// slot table: edge_ids[b*S+s] = node2edge[train_ids[b]*S + s]
// ---------------------------------------------------------------------------
__global__ void k_slot(const int* __restrict__ n2e, const int* __restrict__ train,
                       int* __restrict__ eids) {
  int i = blockIdx.x * 256 + threadIdx.x;       // i < SL
  int b = i >> 4, s = i & 15;
  eids[i] = n2e[(size_t)train[b] * SS + s];
}

// ---------------------------------------------------------------------------
// WMMA GEMM: C[M,Nc] = A[M,K](bf16, lda, optional row gather) @ Wt[Nc,K](bf16)
// The 64-col B panel (contiguous 64*K bf16 in Wt) is staged into LDS once per
// workgroup with the Tensor Data Mover (tensor_load_to_lds + s_wait_tensorcnt),
// then each of the 4 waves computes a 16x64 strip: per k-step 1 A-frag (global,
// software-pipelined) + 4 B-frags (ds_load_b128) + 4 v_wmma.
// A frag (16-bit 16x32): lane<16 row=lane, K {k0..k0+7, k0+16..k0+23}
//                        lane>=16 row=lane-16, K {k0+8.., k0+24..}
// B frag (16-bit 32x16): lane<16 col=lane,    K {k0..k0+7, k0+16..k0+23}
//                        lane>=16 col=lane-16, K {k0+8.., k0+24..}
// C frag: vgpr r -> row r (lanes<16) / row r+8 (lanes>=16), col = lane&15
// ---------------------------------------------------------------------------
__global__ void k_gemm(const __bf16* __restrict__ A, int lda,
                       const int* __restrict__ arows,
                       const __bf16* __restrict__ Wt,
                       int M, int K,
                       float* __restrict__ Cf, int ldcf,
                       __bf16* __restrict__ Cb, int ldcb) {
  extern __shared__ __bf16 Bsm[];              // 64*K bf16 panel
  const int w = threadIdx.x >> 5;
  const int lane = threadIdx.x & 31;
  const int mtile = blockIdx.x * 4 + w;
  const int ntile0 = blockIdx.y * 64;

  // ---- TDM: stage contiguous B panel Wt[ntile0..ntile0+64)[0..K) into LDS ----
  if (w == 0) {
    const unsigned long long ga =
        (unsigned long long)(const void*)(Wt + (size_t)ntile0 * K);
    const unsigned ntot = 64u * (unsigned)K;   // bf16 elements, <= 32768
    v4u g0;
    g0[0] = 1u;                                // count=1, user mode
    g0[1] = 0u;                                // lds_addr = 0 (dynamic LDS base)
    g0[2] = (unsigned)ga;                      // global_addr[31:0]
    g0[3] = (unsigned)((ga >> 32) & 0x1FFFFFFull) | 0x80000000u; // [56:32] | type=2
    v8i g1;
    g1[0] = (int)(1u << 16);                   // data_size=1 (2 bytes), wg_mask=0
    g1[1] = (int)((ntot & 0xFFFFu) << 16);     // tensor_dim0[15:0]
    g1[2] = (int)(((ntot >> 16) & 0xFFFFu) | (1u << 16)); // td0[31:16] | tensor_dim1=1
    g1[3] = (int)((ntot & 0xFFFFu) << 16);     // td1[31:16]=0 | tile_dim0 = ntot
    g1[4] = 0;                                 // tile_dim1=0, tile_dim2=0 (unused)
    g1[5] = (int)ntot;                         // tensor_dim0_stride[31:0]
    g1[6] = 0;
    g1[7] = 0;
    v4i g2 = {0, 0, 0, 0};                     // dims 2/3 unused
    v4i g3 = {0, 0, 0, 0};                     // dim 4 unused
    v8i g4 = {0, 0, 0, 0, 0, 0, 0, 0};         // extra group (6-arg builtin form)
    __builtin_amdgcn_tensor_load_to_lds(g0, g1, g2, g3, g4, 0);
    __builtin_amdgcn_s_wait_tensorcnt(0);
  }
  __syncthreads();
  if (mtile * 16 >= M) return;                 // wave-uniform (after barrier)

  const int mrow = mtile * 16 + (lane & 15);
  const long arow = arows ? (long)arows[mrow] : (long)mrow;
  const __bf16* __restrict__ Ap = A + arow * (long)lda;
  const int hiA = (lane >> 4) * 8;
  const int hiB = (lane >> 4) * 8;

  union AV { v16bf v; uint4 q[2]; };
  AV a_cur, a_nxt;
  a_cur.q[0] = *(const uint4*)(Ap + hiA);
  a_cur.q[1] = *(const uint4*)(Ap + 16 + hiA);

  v8f acc[4] = {};
  for (int k0 = 0; k0 < K; k0 += 32) {
    if (k0 + 32 < K) {                         // software-pipeline next A frag
      a_nxt.q[0] = *(const uint4*)(Ap + k0 + 32 + hiA);
      a_nxt.q[1] = *(const uint4*)(Ap + k0 + 48 + hiA);
    }
#pragma unroll
    for (int nt = 0; nt < 4; ++nt) {
      union { v16bf v; uint4 q[2]; } b;
      const __bf16* bp = Bsm + (size_t)(nt * 16 + (lane & 15)) * K + k0 + hiB;
      b.q[0] = *(const uint4*)(bp);
      b.q[1] = *(const uint4*)(bp + 16);
      acc[nt] = __builtin_amdgcn_wmma_f32_16x16x32_bf16(
          false, a_cur.v, false, b.v, (short)0, acc[nt], false, false);
    }
    if (k0 + 32 < K) a_cur = a_nxt;
  }

  const int rbase = mtile * 16 + ((lane >> 4) ? 8 : 0);
  const int c = lane & 15;
#pragma unroll
  for (int nt = 0; nt < 4; ++nt) {
    const int col = ntile0 + nt * 16 + c;
#pragma unroll
    for (int r = 0; r < 8; ++r) {
      float v = acc[nt][r];
      long row = rbase + r;
      if (Cf) Cf[row * (long)ldcf + col] = v;
      if (Cb) Cb[row * (long)ldcb + col] = (__bf16)v;
    }
  }
}

__device__ __forceinline__ float wave_sum(float v) {
#pragma unroll
  for (int off = 16; off; off >>= 1) v += __shfl_xor(v, off, 32);
  return v;
}
__device__ __forceinline__ float elu1(float x) {
  return x > 0.f ? x : (__expf(x) - 1.f);
}

// ---------------------------------------------------------------------------
// edge attention (layer 0, per gathered slot): softmax over the 2 endpoints
// one wave per (slot, head); lanes cover O=64 (2 elems each)
// ---------------------------------------------------------------------------
__global__ void k_edge_attn(const __bf16* __restrict__ Qe,   // [SL,256]
                            const __bf16* __restrict__ Kn,   // [N,256]
                            const int* __restrict__ eids,    // [SL]
                            const int* __restrict__ adj,     // [E,2] (this mp)
                            __bf16* __restrict__ NE,         // [SL,256]
                            int nslots) {
  const int w = threadIdx.x >> 5, lane = threadIdx.x & 31;
  const int gid = blockIdx.x * 8 + w;
  const int slot = gid >> 2, h = gid & 3;
  if (slot >= nslots) return;
  const int o = lane * 2;
  const long qb = (long)slot * 256 + h * 64 + o;
  const float q0 = (float)Qe[qb], q1 = (float)Qe[qb + 1];
  const int e = eids[slot];
  const int n0 = adj[2 * (long)e], n1 = adj[2 * (long)e + 1];
  const long kb0 = (long)n0 * 256 + h * 64 + o;
  const long kb1 = (long)n1 * 256 + h * 64 + o;
  const float k00 = (float)Kn[kb0], k01 = (float)Kn[kb0 + 1];
  const float k10 = (float)Kn[kb1], k11 = (float)Kn[kb1 + 1];
  float d0 = wave_sum(q0 * k00 + q1 * k01) * 0.125f;   // 1/sqrt(64)
  float d1 = wave_sum(q0 * k10 + q1 * k11) * 0.125f;
  const float m = fmaxf(d0, d1);
  float a0 = __expf(d0 - m), a1 = __expf(d1 - m);
  const float inv = 1.f / (a0 + a1);
  a0 *= inv; a1 *= inv;
  NE[qb]     = (__bf16)elu1(q0 + a0 * k00 + a1 * k10);
  NE[qb + 1] = (__bf16)elu1(q1 + a0 * k01 + a1 * k11);
}

// ---------------------------------------------------------------------------
// node attention (per train node): softmax over S=16 incident-edge K rows
// one wave per (b, head); writes f32 X and bf16 Xb at column offset coff
// ---------------------------------------------------------------------------
__global__ void k_node_attn(const __bf16* __restrict__ Qn,  // [B,256]
                            const __bf16* __restrict__ Ks,  // [B*16,256] slot K
                            float* __restrict__ X,          // [B,512]
                            __bf16* __restrict__ Xb,        // [B,512]
                            int coff, int Bn) {
  __shared__ float lg[8][16];
  const int w = threadIdx.x >> 5, lane = threadIdx.x & 31;
  const int gid = blockIdx.x * 8 + w;
  const int b = gid >> 2, h = gid & 3;
  if (b >= Bn) return;
  const int o = lane * 2;
  const long qb = (long)b * 256 + h * 64 + o;
  const float q0 = (float)Qn[qb], q1 = (float)Qn[qb + 1];
  float k0[16], k1[16];
#pragma unroll
  for (int s = 0; s < 16; ++s) {
    const long kb = ((long)(b * 16 + s)) * 256 + h * 64 + o;
    k0[s] = (float)Ks[kb];
    k1[s] = (float)Ks[kb + 1];
    float d = wave_sum(q0 * k0[s] + q1 * k1[s]);
    if (lane == 0) lg[w][s] = d * 0.125f;
  }
  float mx = -1e30f;
#pragma unroll
  for (int s = 0; s < 16; ++s) mx = fmaxf(mx, lg[w][s]);
  float sum = 0.f, o0 = 0.f, o1 = 0.f;
#pragma unroll
  for (int s = 0; s < 16; ++s) {
    float a = __expf(lg[w][s] - mx);
    sum += a;
    o0 += a * k0[s];
    o1 += a * k1[s];
  }
  const float inv = 1.f / sum;
  o0 = elu1(q0 + o0 * inv);
  o1 = elu1(q1 + o1 * inv);
  const long xb = (long)b * 512 + coff + h * 64 + o;
  X[xb] = o0;       X[xb + 1] = o1;
  Xb[xb] = (__bf16)o0; Xb[xb + 1] = (__bf16)o1;
}

// ---------------------------------------------------------------------------
// gate: softmax over mp of  vg . tanh(G[mp*B+b])
// ---------------------------------------------------------------------------
__global__ void k_gate(const float* __restrict__ G, const float* __restrict__ vg,
                       float* __restrict__ gate, int Bn) {
  const int b = blockIdx.x * 256 + threadIdx.x;
  if (b >= Bn) return;
  float s0 = 0.f, s1 = 0.f;
  for (int e = 0; e < 128; ++e) {
    s0 += tanhf(G[(long)b * 128 + e]) * vg[e];
    s1 += tanhf(G[(long)(Bn + b) * 128 + e]) * vg[e];
  }
  const float m = fmaxf(s0, s1);
  const float e0 = __expf(s0 - m), e1 = __expf(s1 - m);
  const float inv = 1.f / (e0 + e1);
  gate[b] = e0 * inv;
  gate[Bn + b] = e1 * inv;
}

// ---------------------------------------------------------------------------
// pooled = g0*x0 + g1*x1;  logits = pooled @ W_fc + b_fc   (one block per b)
// ---------------------------------------------------------------------------
__global__ void k_pool_fc(const float* __restrict__ X,   // [2*B,512]
                          const float* __restrict__ gate,// [2,B]
                          const float* __restrict__ Wfc, // [512,8]
                          const float* __restrict__ bfc, // [8]
                          float* __restrict__ out, int Bn) {
  __shared__ float pooled[512];
  __shared__ float part[64];
  const int b = blockIdx.x;
  const float g0 = gate[b], g1 = gate[Bn + b];
  for (int d = threadIdx.x; d < 512; d += 256)
    pooled[d] = g0 * X[(long)b * 512 + d] + g1 * X[(long)(Bn + b) * 512 + d];
  __syncthreads();
  if (threadIdx.x < 64) {
    const int c = threadIdx.x & 7, chunk = threadIdx.x >> 3;
    float p = 0.f;
    for (int d = chunk * 64; d < chunk * 64 + 64; ++d) p += pooled[d] * Wfc[d * 8 + c];
    part[threadIdx.x] = p;
  }
  __syncthreads();
  if (threadIdx.x < 8) {
    float s = bfc[threadIdx.x];
    for (int j = 0; j < 8; ++j) s += part[j * 8 + threadIdx.x];
    out[(long)b * 8 + threadIdx.x] = s;
  }
}

// ---------------------------------------------------------------------------
extern "C" void kernel_launch(void* const* d_in, const int* in_sizes, int n_in,
                              void* d_out, int out_size, void* d_ws, size_t ws_size,
                              hipStream_t stream) {
  const float* feats       = (const float*)d_in[0];
  const float* edge_emb    = (const float*)d_in[1];
  const float* W_prep0     = (const float*)d_in[2];
  const float* W_prep1     = (const float*)d_in[3];
  const float* edge_prep_w = (const float*)d_in[4];
  const float* edge_wq_l0  = (const float*)d_in[5];
  const float* edge_wk_l0  = (const float*)d_in[6];
  // d_in[7], d_in[8] (edge_*_l1) are dead code in the reference
  const float* node_wq_l0  = (const float*)d_in[9];
  const float* node_wk_l0  = (const float*)d_in[10];
  const float* node_wq_l1  = (const float*)d_in[11];
  const float* node_wk_l1  = (const float*)d_in[12];
  const float* Wg          = (const float*)d_in[13];
  const float* vg          = (const float*)d_in[14];
  const float* W_fc        = (const float*)d_in[15];
  const float* b_fc        = (const float*)d_in[16];
  const int*   n2e         = (const int*)d_in[17];
  const int*   adj         = (const int*)d_in[18];
  const int*   train_ids   = (const int*)d_in[19];
  float* out = (float*)d_out;           // [B,8] logits then [2,B] gate
  float* gate = out + (size_t)BB * NCLS;

  // ---- workspace carve-up (256B aligned) ----
  char* wp = (char*)d_ws;
  auto alloc = [&](size_t bytes) -> void* {
    void* p = (void*)wp;
    wp += (bytes + 255) & ~(size_t)255;
    return p;
  };
  __bf16* wt_prep0 = (__bf16*)alloc(64 * 128 * 2);
  __bf16* wt_prep1 = (__bf16*)alloc(64 * 128 * 2);
  __bf16* wt_eprep = (__bf16*)alloc((size_t)NMPm * 32 * 128 * 2);
  __bf16* wt_ewq0  = (__bf16*)alloc((size_t)NMPm * HH * 128 * 64 * 2);
  __bf16* wt_ewk0  = (__bf16*)alloc((size_t)NMPm * HH * 128 * 64 * 2);
  __bf16* wt_nwq0  = (__bf16*)alloc((size_t)NMPm * HH * 128 * 64 * 2);
  __bf16* wt_nwk0  = (__bf16*)alloc((size_t)NMPm * HH * 128 * 64 * 2);
  __bf16* wt_nwq1  = (__bf16*)alloc((size_t)NMPm * HH * 256 * 64 * 2);
  __bf16* wt_nwk1  = (__bf16*)alloc((size_t)NMPm * HH * 256 * 64 * 2);
  __bf16* wt_g     = (__bf16*)alloc((size_t)512 * 128 * 2);
  __bf16* featsb   = (__bf16*)alloc((size_t)NN * 64 * 2);
  __bf16* afb      = (__bf16*)alloc((size_t)NN * 128 * 2);   // feats@W_prep1
  __bf16* T0b      = (__bf16*)alloc((size_t)BB * 128 * 2);   // feats[train]@W_prep0
  __bf16* Kn0b     = (__bf16*)alloc((size_t)NN * 256 * 2);
  __bf16* Qn0b     = (__bf16*)alloc((size_t)BB * 256 * 2);
  __bf16* Qn1b     = (__bf16*)alloc((size_t)BB * 256 * 2);
  __bf16* edgeb    = (__bf16*)alloc((size_t)EE * 32 * 2);
  __bf16* aeb      = (__bf16*)alloc((size_t)EE * 128 * 2);   // edge prep
  __bf16* Qe_s     = (__bf16*)alloc((size_t)SL * 256 * 2);   // reused as Ke1_s
  __bf16* Ke0_s    = (__bf16*)alloc((size_t)SL * 256 * 2);
  __bf16* NEb_s    = (__bf16*)alloc((size_t)SL * 256 * 2);
  __bf16* Xb       = (__bf16*)alloc((size_t)2 * BB * 512 * 2);
  float*  X        = (float*)alloc((size_t)2 * BB * 512 * 4);
  float*  G        = (float*)alloc((size_t)2 * BB * 128 * 4);
  int*    eids     = (int*)alloc((size_t)SL * 4);
  (void)ws_size; (void)in_sizes; (void)n_in; (void)out_size;

  auto gemm = [&](const __bf16* A, int lda, const int* rows, const __bf16* Wt,
                  int M, int K, int Nc, float* Cf, int ldcf, __bf16* Cb, int ldcb) {
    dim3 g((M / 16 + 3) / 4, Nc / 64);
    unsigned shmem = (unsigned)(64 * K * 2);   // B panel in LDS (<= 64KB)
    k_gemm<<<g, 128, shmem, stream>>>(A, lda, rows, Wt, M, K, Cf, ldcf, Cb, ldcb);
  };
  auto wt = [&](const float* s, __bf16* d, int K, int Nc, int nmat) {
    int tot = nmat * K * Nc;
    k_wt<<<(tot + 255) / 256, 256, 0, stream>>>(s, d, K, Nc, tot);
  };

  // ---- weight prep (bf16, transposed [Nc,K]) ----
  wt(W_prep0, wt_prep0, 64, 128, 1);
  wt(W_prep1, wt_prep1, 64, 128, 1);
  wt(edge_prep_w, wt_eprep, 32, 128, NMPm);
  wt(edge_wq_l0, wt_ewq0, 128, 64, NMPm * HH);
  wt(edge_wk_l0, wt_ewk0, 128, 64, NMPm * HH);
  wt(node_wq_l0, wt_nwq0, 128, 64, NMPm * HH);
  wt(node_wk_l0, wt_nwk0, 128, 64, NMPm * HH);
  wt(node_wq_l1, wt_nwq1, 256, 64, NMPm * HH);
  wt(node_wk_l1, wt_nwk1, 256, 64, NMPm * HH);
  wt(Wg, wt_g, 512, 128, 1);

  // ---- mp-independent activations ----
  {
    int n = NN * 64;
    k_conv<<<(n + 255) / 256, 256, 0, stream>>>(feats, featsb, n);
  }
  gemm(featsb, 64, nullptr, wt_prep1, NN, 64, 128, nullptr, 0, afb, 128);
  gemm(featsb, 64, train_ids, wt_prep0, BB, 64, 128, nullptr, 0, T0b, 128);

  // ---- per-metapath pipeline ----
  for (int mp = 0; mp < NMPm; ++mp) {
    const __bf16* ewq = wt_ewq0 + (size_t)mp * HH * 64 * 128;  // [256,128]
    const __bf16* ewk = wt_ewk0 + (size_t)mp * HH * 64 * 128;
    const __bf16* nwq0 = wt_nwq0 + (size_t)mp * HH * 64 * 128;
    const __bf16* nwk0 = wt_nwk0 + (size_t)mp * HH * 64 * 128;
    const __bf16* nwq1 = wt_nwq1 + (size_t)mp * HH * 64 * 256; // [256,256]
    const __bf16* nwk1 = wt_nwk1 + (size_t)mp * HH * 64 * 256;
    const int* adj_mp = adj + (size_t)mp * EE * 2;
    float*  X_mp  = X  + (size_t)mp * BB * 512;
    __bf16* Xb_mp = Xb + (size_t)mp * BB * 512;

    { int n = EE * 32;
      k_conv<<<(n + 255) / 256, 256, 0, stream>>>(edge_emb + (size_t)mp * EE * 32, edgeb, n); }
    k_slot<<<SL / 256, 256, 0, stream>>>(n2e + (size_t)mp * NN * SS, train_ids, eids);

    // edge prep + Q/K projections (all WMMA, TDM-staged B panels)
    gemm(edgeb, 32, nullptr, wt_eprep + (size_t)mp * 128 * 32, EE, 32, 128,
         nullptr, 0, aeb, 128);
    gemm(afb, 128, nullptr, ewk, NN, 128, 256, nullptr, 0, Kn0b, 256);
    gemm(aeb, 128, eids, ewq, SL, 128, 256, nullptr, 0, Qe_s, 256);
    gemm(aeb, 128, eids, nwk0, SL, 128, 256, nullptr, 0, Ke0_s, 256);
    gemm(T0b, 128, nullptr, nwq0, BB, 128, 256, nullptr, 0, Qn0b, 256);

    // layer 0: edge attention -> next_edges (slots), node attention -> skip0
    k_edge_attn<<<SL * HH / 8, 256, 0, stream>>>(Qe_s, Kn0b, eids, adj_mp, NEb_s, SL);
    k_node_attn<<<BB * HH / 8, 256, 0, stream>>>(Qn0b, Ke0_s, X_mp, Xb_mp, 0, BB);

    // layer 1: K from updated edges, Q from updated train feats; node attn -> skip1
    gemm(NEb_s, 256, nullptr, nwk1, SL, 256, 256, nullptr, 0, Qe_s /*Ke1_s*/, 256);
    gemm(Xb_mp, 512, nullptr, nwq1, BB, 256, 256, nullptr, 0, Qn1b, 256);
    k_node_attn<<<BB * HH / 8, 256, 0, stream>>>(Qn1b, Qe_s, X_mp, Xb_mp, 256, BB);
  }

  // ---- gated metapath aggregation + classifier ----
  gemm(Xb, 512, nullptr, wt_g, 2 * BB, 512, 128, G, 128, nullptr, 0);
  k_gate<<<(BB + 255) / 256, 256, 0, stream>>>(G, vg, gate, BB);
  k_pool_fc<<<BB, 256, 0, stream>>>(X, gate, W_fc, b_fc, out, BB);
}
